// ELiCiT_50087908606687
// MI455X (gfx1250) — matrix operand order repaired and checked by reading, not compiled, not run on hardware
//
#include <hip/hip_runtime.h>

typedef _Float16 half_t;
typedef __attribute__((ext_vector_type(16))) _Float16 v16h;
typedef __attribute__((ext_vector_type(8)))  float    v8f;

#define DIM0  50000
#define DIM1  50000
#define SROWS (DIM0 + DIM1)
#define NF    64
#define NQ    16
#define NNZ_  1000000

__device__ __forceinline__ float sigmoidf_(float x) {
    return 1.0f / (1.0f + __expf(-x));
}

// ---------------- K0: zero a float region (accumulator + counters) ------------
__global__ void k_zero(float* __restrict__ ptr, int n) {
    int i = blockIdx.x * blockDim.x + threadIdx.x;
    int stride = gridDim.x * blockDim.x;
    for (; i < n; i += stride) ptr[i] = 0.0f;
}

// ---------------- K1: quantize feats/ifeats against sigmoid codebooks --------
// 256 threads/block: 4 rows x 64 feats. Codebooks (2 axes x 64 x 16) in LDS.
__global__ void k_quant(const float* __restrict__ feats,
                        const float* __restrict__ ifeats,
                        const float* __restrict__ keys,
                        const float* __restrict__ ikeys,
                        float* __restrict__ pOut,
                        float* __restrict__ ipOut) {
    __shared__ float sk [2 * NF * NQ];
    __shared__ float sik[2 * NF * NQ];
    int tid = threadIdx.x;
    for (int i = tid; i < 2 * NF * NQ; i += blockDim.x) {
        sk [i] = sigmoidf_(keys[i]);
        sik[i] = sigmoidf_(ikeys[i]);
    }
    __syncthreads();

    int r = blockIdx.x * 4 + (tid >> 6);
    if (r >= SROWS) return;
    int f    = tid & 63;
    int axis = (r >= DIM0) ? 1 : 0;
    const float* cb  = &sk [axis * NF * NQ + f * NQ];
    const float* icb = &sik[axis * NF * NQ + f * NQ];
    size_t off = (size_t)r * NF + f;
    {
        float tf = sigmoidf_(feats[off]);
        float best = cb[0];
        float bd   = fabsf(tf - best);
        #pragma unroll
        for (int q = 1; q < NQ; ++q) {
            float d = fabsf(tf - cb[q]);
            if (d < bd) { bd = d; best = cb[q]; }   // first-min, matches argmin
        }
        pOut[off] = best;
    }
    {
        float tf = sigmoidf_(ifeats[off]);
        float best = icb[0];
        float bd   = fabsf(tf - best);
        #pragma unroll
        for (int q = 1; q < NQ; ++q) {
            float d = fabsf(tf - icb[q]);
            if (d < bd) { bd = d; best = icb[q]; }
        }
        ipOut[off] = best;
    }
}

// ---------------- K2: per-row degree counts (float, L2 atomics) ---------------
__global__ void k_count(const long long* __restrict__ i0,
                        const long long* __restrict__ i1,
                        float* __restrict__ cnt0,
                        float* __restrict__ cnt1) {
    int n = blockIdx.x * blockDim.x + threadIdx.x;
    if (n >= NNZ_) return;
    unsafeAtomicAdd(&cnt0[(int)i0[n]], 1.0f);
    unsafeAtomicAdd(&cnt1[(int)i1[n]], 1.0f);
}

// ---------------- K3: scatter (ip[partner]-0.5) into acc ----------------------
// 32 threads per edge, 2 features each (float2 loads, 4 f32 atomics/thread).
__global__ void k_scatter(const long long* __restrict__ i0,
                          const long long* __restrict__ i1,
                          const float* __restrict__ ip,
                          float* __restrict__ acc) {
    long long gid = (long long)blockIdx.x * blockDim.x + threadIdx.x;
    int n = (int)(gid >> 5);
    if (n >= NNZ_) return;
    int f  = ((int)(gid & 31)) * 2;
    int r0 = (int)i0[n];
    int r1 = (int)i1[n];
    size_t o0 = (size_t)r0 * NF + f;
    size_t o1 = (size_t)(DIM0 + r1) * NF + f;
    float2 v1 = *(const float2*)&ip[o1];   // partner of mode-0 row
    float2 v0 = *(const float2*)&ip[o0];   // partner of mode-1 row
    unsafeAtomicAdd(&acc[o0],     v1.x - 0.5f);
    unsafeAtomicAdd(&acc[o0 + 1], v1.y - 0.5f);
    unsafeAtomicAdd(&acc[o1],     v0.x - 0.5f);
    unsafeAtomicAdd(&acc[o1 + 1], v0.y - 0.5f);
}

// ---------------- K4: build f16 SDDMM operands P (mode 0) and Q (mode 1) ------
// P[r, f]    = a0 = p + q          Q[r, f]    = scale*(V0*b0 + V1*b1)
// P[r, 64+f] = a1 = 1 - p - q      Q[r, 64+f] = scale*(V2*b0 + V3*b1)
__global__ void k_build(const float* __restrict__ p,
                        const float* __restrict__ acc,
                        const float* __restrict__ cnt0,
                        const float* __restrict__ cnt1,
                        const float* __restrict__ values,
                        const float* __restrict__ scale,
                        half_t* __restrict__ Pm,
                        half_t* __restrict__ Qm) {
    int gid = blockIdx.x * blockDim.x + threadIdx.x;
    if (gid >= DIM0 * NF) return;
    int r = gid >> 6;
    int f = gid & 63;
    float sc = scale[0];
    {   // mode-0 row r
        float rn  = rsqrtf(cnt0[r] + 1e-12f);
        size_t off = (size_t)r * NF + f;
        float q  = acc[off] * rn;
        float pv = p[off];
        Pm[(size_t)r * 128 + f]      = (half_t)(pv + q);
        Pm[(size_t)r * 128 + 64 + f] = (half_t)((1.0f - pv) - q);
    }
    {   // mode-1 row (DIM0 + r)
        float rn  = rsqrtf(cnt1[r] + 1e-12f);
        size_t off = (size_t)(DIM0 + r) * NF + f;
        float q  = acc[off] * rn;
        float pv = p[off];
        float b0 = pv + q;
        float b1 = (1.0f - pv) - q;
        Qm[(size_t)r * 128 + f]      = (half_t)(sc * (values[0 * NF + f] * b0 + values[1 * NF + f] * b1));
        Qm[(size_t)r * 128 + 64 + f] = (half_t)(sc * (values[2 * NF + f] * b0 + values[3 * NF + f] * b1));
    }
}

// ---------------- K5: SDDMM edge kernel via v_wmma_f32_16x16x32_f16 -----------
// One wave32 per 16-edge tile: C(16x16) = A(16x128) * B(128x16) in 4 WMMA
// steps (K=32 each); out = diag(C).
// A layout (16-bit A 16x32): lanes 0-15 hold M=lane, K = kb+{0..7,16..23};
//                            lanes 16-31 hold M=lane-16, K = kb+{8..15,24..31}.
// B layout (16-bit B 32x16): lanes 0-15 hold N=lane, K = kb+{0..7,16..23};
//                            lanes 16-31 hold N=lane-16, K = kb+{8..15,24..31}.
__global__ void k_edge(const long long* __restrict__ i0,
                       const long long* __restrict__ i1,
                       const half_t* __restrict__ Pm,
                       const half_t* __restrict__ Qm,
                       float* __restrict__ out) {
    const int ntiles = NNZ_ / 16;
    int tile = blockIdx.x * (blockDim.x >> 5) + (threadIdx.x >> 5);
    if (tile >= ntiles) return;              // wave-uniform: EXEC stays all-1s
    int lane = threadIdx.x & 31;
    int m    = lane & 15;
    int hi   = lane >> 4;
    int base = tile * 16;

    int r0 = (int)i0[base + m];
    int r1 = (int)i1[base + m];
    const half_t* prow = Pm + (size_t)r0 * 128;
    const half_t* qrow = Qm + (size_t)r1 * 128;

    v8f c = {};
    #pragma unroll
    for (int kk = 0; kk < 4; ++kk) {
        int kb = kk * 32;
        union { v16h v; int4 q[2]; } ua, ub;
        // A: two 16B chunks per lane
        ua.q[0] = *(const int4*)(prow + kb + hi * 8);
        ua.q[1] = *(const int4*)(prow + kb + 16 + hi * 8);
        // B: two 16B chunks per lane
        const half_t* bs = qrow + kb + hi * 8;
        ub.q[0] = *(const int4*)(bs);
        ub.q[1] = *(const int4*)(bs + 16);
        c = __builtin_amdgcn_wmma_f32_16x16x32_f16(
                /*neg_a=*/false, ua.v, /*neg_b=*/false, ub.v,
                /*c_mod=*/(short)0, c, /*reuse_a=*/false, /*reuse_b=*/false);
    }

    // diag(C): lane m (m<8) has C[m][m] in c[m]; lane m+16 (m>=8) in c[m-8].
    int sel = (lane < 8) ? lane : (lane - 24);
    float d = c[0];
    d = (sel == 1) ? c[1] : d;
    d = (sel == 2) ? c[2] : d;
    d = (sel == 3) ? c[3] : d;
    d = (sel == 4) ? c[4] : d;
    d = (sel == 5) ? c[5] : d;
    d = (sel == 6) ? c[6] : d;
    d = (sel == 7) ? c[7] : d;
    bool doStore = (lane < 8) || (lane >= 24);
    if (doStore) out[base + ((lane < 8) ? lane : (lane - 16))] = d;
}

// ------------------------------- launcher -------------------------------------
extern "C" void kernel_launch(void* const* d_in, const int* in_sizes, int n_in,
                              void* d_out, int out_size, void* d_ws, size_t ws_size,
                              hipStream_t stream) {
    (void)in_sizes; (void)n_in; (void)out_size; (void)ws_size;
    const float*     feats  = (const float*)d_in[0];
    const float*     ifeats = (const float*)d_in[1];
    const float*     keys   = (const float*)d_in[2];
    const float*     ikeys  = (const float*)d_in[3];
    const float*     values = (const float*)d_in[4];
    const float*     scale  = (const float*)d_in[5];
    const long long* idxs   = (const long long*)d_in[6];   // (2, NNZ) int64
    const long long* i0 = idxs;
    const long long* i1 = idxs + NNZ_;
    float* out = (float*)d_out;

    // workspace layout (acc, cnt0, cnt1 contiguous so one zero pass covers them)
    float* p    = (float*)d_ws;                       // SROWS*NF
    float* ip   = p   + (size_t)SROWS * NF;           // SROWS*NF
    float* acc  = ip  + (size_t)SROWS * NF;           // SROWS*NF
    float* cnt0 = acc + (size_t)SROWS * NF;           // DIM0
    float* cnt1 = cnt0 + DIM0;                        // DIM1
    half_t* Pm  = (half_t*)(cnt1 + DIM1);             // DIM0*128 halves
    half_t* Qm  = Pm + (size_t)DIM0 * 128;            // DIM1*128 halves

    int zn = SROWS * NF + DIM0 + DIM1;
    hipLaunchKernelGGL(k_zero, dim3(1024), dim3(256), 0, stream, acc, zn);
    hipLaunchKernelGGL(k_quant, dim3((SROWS + 3) / 4), dim3(256), 0, stream,
                       feats, ifeats, keys, ikeys, p, ip);
    hipLaunchKernelGGL(k_count, dim3((NNZ_ + 255) / 256), dim3(256), 0, stream,
                       i0, i1, cnt0, cnt1);
    hipLaunchKernelGGL(k_scatter, dim3(NNZ_ * 32 / 256), dim3(256), 0, stream,
                       i0, i1, ip, acc);
    hipLaunchKernelGGL(k_build, dim3(DIM0 * NF / 256), dim3(256), 0, stream,
                       p, acc, cnt0, cnt1, values, scale, Pm, Qm);
    const int ntiles = NNZ_ / 16;                     // 62500, exact
    const int wavesPerBlock = 8;                      // 256 threads
    hipLaunchKernelGGL(k_edge, dim3((ntiles + wavesPerBlock - 1) / wavesPerBlock),
                       dim3(256), 0, stream, i0, i1, Pm, Qm, out);
}
